// FFNMoE_21955872817238
// MI455X (gfx1250) — matrix operand order repaired and checked by reading, not compiled, not run on hardware
//
#include <hip/hip_runtime.h>
#include <hip/hip_bf16.h>
#include <cstdint>

// ---------------------------------------------------------------------------
// MoE FFN (gate + top-2 routed experts + shared expert), fused, bf16 WMMA.
// B=8 S=4096 H=512 E=8 K=2 DF=128 SH=512  ->  T=32768 tokens.
// ---------------------------------------------------------------------------

typedef __bf16 bf16;
typedef __attribute__((ext_vector_type(8)))  __bf16 v8bf;
typedef __attribute__((ext_vector_type(16))) __bf16 v16bf;
typedef __attribute__((ext_vector_type(8)))  float  v8f;

#define T_TOK 32768
#define HDIM  512
#define EDIM  8
#define DFDIM 128
#define SHDIM 512

#define ROWS  64      // tokens per workgroup tile
#define NWAVE 4       // waves per workgroup (wave32)
#define XS_LD 520     // bf16 row stride for 512-wide LDS tiles (+8 pad -> 4-bank rotation)
#define H1_LD 136     // bf16 row stride for 128-wide LDS tiles

// bf16 transposed-weight workspace layout (element offsets)
#define W1N   (EDIM * DFDIM * HDIM)   // w1t[e][f][h]
#define W2N   (EDIM * HDIM * DFDIM)   // w2t[e][h][f]
#define SW1N  (SHDIM * HDIM)          // sw1t[n][h]
#define SW2N  (HDIM * SHDIM)          // sw2t[n][k]
#define TOTW  (W1N + W2N + SW1N + SW2N)

// LDS: xs[64][520] + s1[64][520] + h1[8][64][136] (bf16) + comb[64][8] (f32)
#define SMEM_BYTES ((2 * ROWS * XS_LD + EDIM * ROWS * H1_LD) * 2 + ROWS * EDIM * 4)

// ---------------------------------------------------------------------------
// Fragment helpers (CDNA5 wave32 WMMA 16x16x32 bf16 layouts, CDNA5 ISA)
// ---------------------------------------------------------------------------
__device__ __forceinline__ v16bf make_frag(v8bf lo, v8bf hi) {
  v16bf r;
#pragma unroll
  for (int i = 0; i < 8; ++i) { r[i] = lo[i]; r[8 + i] = hi[i]; }
  return r;
}

// A (16x32, row-major source, ld in bf16 elems). p points at (row0, k0).
// lanes 0-15: m=lane, v0-3 K=0..7, v4-7 K=16..23 ; lanes 16-31: +8 on K.
__device__ __forceinline__ v16bf load_a(const bf16* p, int ld, int lane) {
  const int m  = lane & 15;
  const int hh = lane >> 4;
  const bf16* q = p + m * ld + 8 * hh;
  v8bf lo = *(const v8bf*)(q);        // K = 8h .. 8h+7
  v8bf hi = *(const v8bf*)(q + 16);   // K = 16+8h .. 16+8h+7
  return make_frag(lo, hi);
}

// B (32x16) from [N][K] transposed weights, ldk in bf16 elems. p points at (n0, k0).
// lanes 0-15: n=lane, v0-3 K=0..7, v4-7 K=16..23 ; lanes 16-31: +8 on K.
__device__ __forceinline__ v16bf load_b(const bf16* p, int ldk, int lane) {
  const int n  = lane & 15;
  const int hh = lane >> 4;
  const bf16* q = p + n * ldk + 8 * hh;
  v8bf lo = *(const v8bf*)(q);        // K = 8h .. 8h+7
  v8bf hi = *(const v8bf*)(q + 16);   // K = 16+8h .. 16+8h+7
  return make_frag(lo, hi);
}

__device__ __forceinline__ v8f wmma_bf16(v16bf a, v16bf b, v8f c) {
  return __builtin_amdgcn_wmma_f32_16x16x32_bf16(false, a, false, b, (short)0, c,
                                                 false, false);
}

// ---------------------------------------------------------------------------
// Prep: fp32 weights -> bf16, transposed so WMMA B-fragments are contiguous in K.
// ---------------------------------------------------------------------------
__global__ __launch_bounds__(256) void moe_prep_kernel(
    const float* __restrict__ w1, const float* __restrict__ w2,
    const float* __restrict__ sw1, const float* __restrict__ sw2,
    bf16* __restrict__ w1t, bf16* __restrict__ w2t,
    bf16* __restrict__ sw1t, bf16* __restrict__ sw2t) {
  int i = blockIdx.x * blockDim.x + threadIdx.x;
  if (i < W1N) {                                  // w1t[e][f][h] = w1[e][h][f]
    int e = i / (DFDIM * HDIM);
    int r = i - e * DFDIM * HDIM;
    int f = r / HDIM, h = r - f * HDIM;
    w1t[i] = (bf16)w1[(size_t)e * HDIM * DFDIM + (size_t)h * DFDIM + f];
  } else if (i < W1N + W2N) {                     // w2t[e][h][f] = w2[e][f][h]
    int j = i - W1N;
    int e = j / (HDIM * DFDIM);
    int r = j - e * HDIM * DFDIM;
    int h = r / DFDIM, f = r - h * DFDIM;
    w2t[j] = (bf16)w2[(size_t)e * DFDIM * HDIM + (size_t)f * HDIM + h];
  } else if (i < W1N + W2N + SW1N) {              // sw1t[n][h] = sw1[h][n]
    int j = i - W1N - W2N;
    int n = j / HDIM, h = j - n * HDIM;
    sw1t[j] = (bf16)sw1[(size_t)h * SHDIM + n];
  } else if (i < TOTW) {                          // sw2t[n][k] = sw2[k][n]
    int j = i - W1N - W2N - SW1N;
    int n = j / SHDIM, k = j - n * SHDIM;
    sw2t[j] = (bf16)sw2[(size_t)k * HDIM + n];
  }
}

// ---------------------------------------------------------------------------
// Fused main kernel: 1 workgroup = 64 tokens, 4 waves, each wave a 16-row band.
// ---------------------------------------------------------------------------
__global__ __launch_bounds__(NWAVE * 32) void moe_main_kernel(
    const float* __restrict__ x,      // [T][H]
    const float* __restrict__ gate_w, // [E][H]
    const float* __restrict__ b1,     // [E][DF]
    const float* __restrict__ b2,     // [E][H]
    const float* __restrict__ sb1,    // [SH]
    const float* __restrict__ sb2,    // [H]
    const bf16* __restrict__ w1t,     // [E][DF][H]
    const bf16* __restrict__ w2t,     // [E][H][DF]
    const bf16* __restrict__ sw1t,    // [SH][H]
    const bf16* __restrict__ sw2t,    // [H][SH]
    float* __restrict__ out) {        // [T][H]
  extern __shared__ char smem[];
  bf16*  xs   = (bf16*)smem;                         // [ROWS][XS_LD]
  bf16*  s1   = xs + ROWS * XS_LD;                   // [ROWS][XS_LD]
  bf16*  h1   = s1 + ROWS * XS_LD;                   // [E][ROWS][H1_LD]
  float* comb = (float*)(h1 + EDIM * ROWS * H1_LD);  // [ROWS][E]

  const int tid  = threadIdx.x;
  const int wave = tid >> 5;
  const int lane = tid & 31;
  const int band = wave * 16;                 // this wave's rows within the tile
  const int row0 = blockIdx.x * ROWS;         // tile base token

  // ---- Phase 1: cooperative load x tile, fp32 -> bf16 into LDS ------------
  for (int i = tid; i < ROWS * (HDIM / 4); i += blockDim.x) {
    int r  = i >> 7;            // / 128
    int c4 = i & 127;
    float4 v = ((const float4*)x)[(size_t)(row0 + r) * (HDIM / 4) + c4];
    bf16* d = xs + r * XS_LD + c4 * 4;
    d[0] = (bf16)v.x; d[1] = (bf16)v.y; d[2] = (bf16)v.z; d[3] = (bf16)v.w;
  }
  __syncthreads();

  // ---- Phase 2: gate (softmax over E=8, top-2, normalized combine) --------
  if (lane < 16) {
    int r = band + lane;
    const bf16* xr = xs + r * XS_LD;
    float lg[EDIM];
#pragma unroll
    for (int e = 0; e < EDIM; ++e) {
      const float* gw = gate_w + (size_t)e * HDIM;
      float acc = 0.f;
      for (int h = 0; h < HDIM; ++h) acc += (float)xr[h] * gw[h];
      lg[e] = acc;
    }
    float mx = lg[0];
#pragma unroll
    for (int e = 1; e < EDIM; ++e) mx = lg[e] > mx ? lg[e] : mx;
    float sum = 0.f, p[EDIM];
#pragma unroll
    for (int e = 0; e < EDIM; ++e) { p[e] = __expf(lg[e] - mx); sum += p[e]; }
    float inv = 1.f / sum;
#pragma unroll
    for (int e = 0; e < EDIM; ++e) p[e] *= inv;
    int i1 = 0; float p1 = p[0];
#pragma unroll
    for (int e = 1; e < EDIM; ++e) if (p[e] > p1) { p1 = p[e]; i1 = e; }
    int i2 = -1; float p2 = -1.f;
#pragma unroll
    for (int e = 0; e < EDIM; ++e)
      if (e != i1 && p[e] > p2) { p2 = p[e]; i2 = e; }
    float winv = 1.f / (p1 + p2 + 1e-20f);
#pragma unroll
    for (int e = 0; e < EDIM; ++e)
      comb[r * EDIM + e] = (e == i1) ? p1 * winv : ((e == i2) ? p2 * winv : 0.f);
  }
  __syncthreads();

  // ---- Phase 3: routed fc1, all experts: h1[e] = relu(X @ w1[e] + b1[e]) --
  for (int e = 0; e < EDIM; ++e) {
    const bf16* w1e = w1t + (size_t)e * DFDIM * HDIM;
    for (int f0 = 0; f0 < DFDIM; f0 += 16) {
      v8f acc = {};
      for (int k0 = 0; k0 < HDIM; k0 += 32) {
        __builtin_prefetch(w1e + (size_t)f0 * HDIM + k0 + 64, 0, 1);
        v16bf a = load_a(xs + band * XS_LD + k0, XS_LD, lane);
        v16bf b = load_b(w1e + (size_t)f0 * HDIM + k0, HDIM, lane);
        acc = wmma_bf16(a, b, acc);
      }
      const int n  = f0 + (lane & 15);
      const int hh = lane >> 4;
      const float bv = b1[e * DFDIM + n];
#pragma unroll
      for (int r = 0; r < 8; ++r) {
        float v = acc[r] + bv;
        v = v > 0.f ? v : 0.f;
        h1[((size_t)e * ROWS + band + 8 * hh + r) * H1_LD + n] = (bf16)v;
      }
    }
  }

  // ---- Phase 4: shared fc1: s1 = relu(X @ sw1 + sb1) ----------------------
  for (int n0 = 0; n0 < SHDIM; n0 += 16) {
    v8f acc = {};
    for (int k0 = 0; k0 < HDIM; k0 += 32) {
      __builtin_prefetch(sw1t + (size_t)n0 * HDIM + k0 + 64, 0, 1);
      v16bf a = load_a(xs + band * XS_LD + k0, XS_LD, lane);
      v16bf b = load_b(sw1t + (size_t)n0 * HDIM + k0, HDIM, lane);
      acc = wmma_bf16(a, b, acc);
    }
    const int n  = n0 + (lane & 15);
    const int hh = lane >> 4;
    const float bv = sb1[n];
#pragma unroll
    for (int r = 0; r < 8; ++r) {
      float v = acc[r] + bv;
      v = v > 0.f ? v : 0.f;
      s1[(band + 8 * hh + r) * XS_LD + n] = (bf16)v;
    }
  }

  // ---- Phase 5: output = s1 @ sw2 + sb2 + sum_e comb[:,e]*(h1[e] @ w2[e] + b2[e])
  for (int n0 = 0; n0 < HDIM; n0 += 16) {
    v8f accS = {};
    for (int k0 = 0; k0 < SHDIM; k0 += 32) {
      v16bf a = load_a(s1 + band * XS_LD + k0, XS_LD, lane);
      v16bf b = load_b(sw2t + (size_t)n0 * SHDIM + k0, SHDIM, lane);
      accS = wmma_bf16(a, b, accS);
    }
    const int n  = n0 + (lane & 15);
    const int hh = lane >> 4;
    float ov[8];
#pragma unroll
    for (int r = 0; r < 8; ++r) ov[r] = accS[r] + sb2[n];

    for (int e = 0; e < EDIM; ++e) {
      v8f accE = {};
      const bf16* w2e = w2t + (size_t)e * HDIM * DFDIM;
#pragma unroll
      for (int k0 = 0; k0 < DFDIM; k0 += 32) {
        v16bf a = load_a(h1 + ((size_t)e * ROWS + band) * H1_LD + k0, H1_LD, lane);
        v16bf b = load_b(w2e + (size_t)n0 * DFDIM + k0, DFDIM, lane);
        accE = wmma_bf16(a, b, accE);
      }
      const float b2v = b2[e * HDIM + n];
#pragma unroll
      for (int r = 0; r < 8; ++r) {
        float cw = comb[(band + 8 * hh + r) * EDIM + e];
        ov[r] += cw * (accE[r] + b2v);
      }
    }
#pragma unroll
    for (int r = 0; r < 8; ++r)
      out[(size_t)(row0 + band + 8 * hh + r) * HDIM + n] = ov[r];
  }
}

// ---------------------------------------------------------------------------
extern "C" void kernel_launch(void* const* d_in, const int* in_sizes, int n_in,
                              void* d_out, int out_size, void* d_ws, size_t ws_size,
                              hipStream_t stream) {
  const float* x      = (const float*)d_in[0];
  const float* gate_w = (const float*)d_in[1];
  const float* w1     = (const float*)d_in[2];
  const float* b1     = (const float*)d_in[3];
  const float* w2     = (const float*)d_in[4];
  const float* b2     = (const float*)d_in[5];
  const float* sw1    = (const float*)d_in[6];
  const float* sb1    = (const float*)d_in[7];
  const float* sw2    = (const float*)d_in[8];
  const float* sb2    = (const float*)d_in[9];
  float* out = (float*)d_out;

  bf16* w1t  = (bf16*)d_ws;
  bf16* w2t  = w1t + W1N;
  bf16* sw1t = w2t + W2N;
  bf16* sw2t = sw1t + SW1N;

  (void)hipFuncSetAttribute((const void*)moe_main_kernel,
                            hipFuncAttributeMaxDynamicSharedMemorySize,
                            SMEM_BYTES);

  moe_prep_kernel<<<(TOTW + 255) / 256, 256, 0, stream>>>(
      w1, w2, sw1, sw2, w1t, w2t, sw1t, sw2t);

  moe_main_kernel<<<T_TOK / ROWS, NWAVE * 32, SMEM_BYTES, stream>>>(
      x, gate_w, b1, b2, sb1, sb2, w1t, w2t, sw1t, sw2t, out);

  (void)in_sizes; (void)n_in; (void)out_size; (void)ws_size;
}
